// CausalSelfAttention_75857712382575
// MI455X (gfx1250) — matrix operand order repaired and checked by reading, not compiled, not run on hardware
//
#include <hip/hip_runtime.h>
#include <hip/hip_bf16.h>
#include <stdint.h>

typedef __bf16 bf16;
typedef bf16  v16bf __attribute__((ext_vector_type(16)));
typedef bf16  v8bf  __attribute__((ext_vector_type(8)));
typedef bf16  v4bf  __attribute__((ext_vector_type(4)));
typedef float v8f   __attribute__((ext_vector_type(8)));
typedef unsigned int tdm_v4u __attribute__((ext_vector_type(4)));
typedef int          tdm_v8i __attribute__((ext_vector_type(8)));
typedef int          tdm_v4i __attribute__((ext_vector_type(4)));

// Problem constants (match reference)
#define NB   2
#define TT   2048
#define CC   1024
#define NH   16
#define DD   64
#define N3   3072
#define MM   4096   // NB*TT

__device__ __forceinline__ bf16 to_bf16(float f) { return (bf16)f; }

__device__ __forceinline__ v8f wmma_bf16(v16bf a, v16bf b, v8f c) {
  return __builtin_amdgcn_wmma_f32_16x16x32_bf16(false, a, false, b, (short)0, c,
                                                 false, false);
}

__device__ __forceinline__ v8bf ld8(const bf16* p) { return *(const v8bf*)p; }

__device__ __forceinline__ v16bf join16(v8bf a, v8bf b) {
  return __builtin_shufflevector(a, b, 0, 1, 2, 3, 4, 5, 6, 7, 8, 9, 10, 11, 12,
                                 13, 14, 15);
}

// A-matrix 16x32 bf16 fragment from row-major [16][ld] (contiguous k order used by this file, not the ISA table order):
// lanes 0-15: M=lane, K=0..15; lanes 16-31: M=lane-16, K=16..31
__device__ __forceinline__ v16bf load_frag_a(const bf16* p, int ld) {
  const int lane = threadIdx.x & 31;
  const int h = lane >> 4, m = lane & 15;
  const bf16* row = p + m * ld + h * 16;
  return join16(ld8(row), ld8(row + 8));
}

// B fragment where memory holds B^T: element (k,n) at p[n*ld + k]
__device__ __forceinline__ v16bf load_frag_bT(const bf16* p, int ld) {
  const int lane = threadIdx.x & 31;
  const int kb = (lane >> 4) * 16, n = lane & 15;
  const bf16* row = p + n * ld + kb;
  return join16(ld8(row), ld8(row + 8));
}

// B fragment from LDS pre-swizzled layout: [(nc*32 + lane)*24 + i]
__device__ __forceinline__ v16bf load_frag_b_sw(const bf16* p) {
  const int lane = threadIdx.x & 31;
  const bf16* q = p + lane * 24;
  return join16(ld8(q), ld8(q + 8));
}

// TDM: async 2D tile load global->LDS (bf16, tile 32 x nrows, row stride
// src_ld elements, LDS row stride 40 bf16 via pad: 4 dwords after every 16).
__device__ __forceinline__ void tdm_load_tile_bf16(
    unsigned lds_off, const bf16* gsrc, int nrows, int src_ld) {
  const uint64_t ga = (uint64_t)(uintptr_t)gsrc;
  tdm_v4u g0;
  g0[0] = 1u;                                   // count=1, user descriptor
  g0[1] = lds_off;                              // lds_addr (bytes)
  g0[2] = (unsigned)(ga & 0xFFFFFFFFu);         // global_addr[31:0]
  g0[3] = (unsigned)((ga >> 32) & 0x1FFFFFFu)   // global_addr[56:32]
          | (2u << 30);                         // type = 2 ("image")
  tdm_v8i g1;
  g1[0] = (1 << 16)                             // data_size = 1 -> 2 bytes
        | (1 << 20)                             // pad_enable
        | (3 << 22)                             // pad_interval: 16 dwords
        | (3 << 25);                            // pad_amount: 4 dwords
  g1[1] = (32 & 0xFFFF) << 16;                  // tensor_dim0 = 32 (lo16)
  g1[2] = (nrows & 0xFFFF) << 16;               // dim0 hi16=0 | tensor_dim1 lo
  g1[3] = (32 << 16);                           // dim1 hi16=0 | tile_dim0 = 32
  g1[4] = (nrows & 0xFFFF);                     // tile_dim1 = nrows, tile_dim2=0
  g1[5] = src_ld;                               // tensor_dim0_stride[31:0]
  g1[6] = 0;                                    // stride0 hi | stride1 lo
  g1[7] = 0;
  tdm_v4i gz = {0, 0, 0, 0};                    // 2D tensor: groups 2/3 unused
#if defined(__clang_major__) && (__clang_major__ >= 23)
  tdm_v8i gz8 = {0, 0, 0, 0, 0, 0, 0, 0};
  __builtin_amdgcn_tensor_load_to_lds(g0, g1, gz, gz, gz8, 0);
#else
  __builtin_amdgcn_tensor_load_to_lds(g0, g1, gz, gz, 0);
#endif
}

// ---------------------------------------------------------------------------
// Kernel 1: qkv = x @ w_attn + b_attn -> bf16 Q (scaled), K ([B][H][T][D]),
// V transposed ([B][H][D][T]). Block 128 thr (4 waves), tile 128M x 64N x 32K.
// ---------------------------------------------------------------------------
__global__ __launch_bounds__(128) void qkv_kernel(
    const float* __restrict__ x, const float* __restrict__ w,
    const float* __restrict__ bias, bf16* __restrict__ Qb,
    bf16* __restrict__ Kb, bf16* __restrict__ Vt) {
  __shared__ alignas(16) bf16 As[128 * 40];
  __shared__ alignas(16) bf16 Bsw[4 * 32 * 24];
  const int tid = threadIdx.x, wave = tid >> 5, lane = tid & 31;
  const int h = lane >> 4, lo = lane & 15;
  const int gn0 = blockIdx.x * 64;
  const int gm0 = blockIdx.y * 128;

  v8f acc[2][4] = {};
  for (int k0 = 0; k0 < CC; k0 += 32) {
    // stage A 128x32 (float4 loads -> bf16)
#pragma unroll
    for (int i = 0; i < 8; ++i) {
      int idx = tid + i * 128;
      int r = idx >> 3, c = (idx & 7) * 4;
      const float4 f = *(const float4*)&x[(size_t)(gm0 + r) * CC + k0 + c];
      v4bf t;
      t[0] = to_bf16(f.x); t[1] = to_bf16(f.y);
      t[2] = to_bf16(f.z); t[3] = to_bf16(f.w);
      *(v4bf*)&As[r * 40 + c] = t;
    }
    // stage B 32x64 swizzled into fragment order
#pragma unroll
    for (int i = 0; i < 4; ++i) {
      int idx = tid + i * 128;
      int r = idx >> 4, c0 = (idx & 15) * 4;
      const float4 f = *(const float4*)&w[(size_t)(k0 + r) * N3 + gn0 + c0];
      int base = (((c0 >> 4) * 32) + ((r >> 4) * 16) + (c0 & 15)) * 24 + (r & 15);
      Bsw[base +  0] = to_bf16(f.x);
      Bsw[base + 24] = to_bf16(f.y);
      Bsw[base + 48] = to_bf16(f.z);
      Bsw[base + 72] = to_bf16(f.w);
    }
    if (k0 + 32 < CC) {  // pull next K-tile toward the caches
      __builtin_prefetch(&x[(size_t)(gm0 + (tid >> 3)) * CC + k0 + 32], 0, 3);
      __builtin_prefetch(&w[(size_t)(k0 + 32 + (tid >> 2)) * N3 + gn0], 0, 3);
    }
    __syncthreads();
    v16bf a0 = load_frag_a(As + (wave * 32 + 0) * 40, 40);
    v16bf a1 = load_frag_a(As + (wave * 32 + 16) * 40, 40);
#pragma unroll
    for (int nc = 0; nc < 4; ++nc) {
      v16bf b = load_frag_b_sw(Bsw + nc * 32 * 24);
      acc[0][nc] = wmma_bf16(a0, b, acc[0][nc]);
      acc[1][nc] = wmma_bf16(a1, b, acc[1][nc]);
    }
    __syncthreads();
  }
  // Epilogue. sel/head/batch are uniform for the whole block.
  const int sel = gn0 >> 10;            // 0=q 1=k 2=v
  const int head = (gn0 & 1023) >> 6;
  const int b_ = gm0 >> 11;
  const int t0 = (gm0 & 2047) + wave * 32;
  const size_t bh = (size_t)b_ * NH + head;

  if (sel == 2) {
    // V transposed: each lane's 8 j-values contiguous in t -> b128 stores
#pragma unroll
    for (int nc = 0; nc < 4; ++nc) {
      const int d = nc * 16 + lo;
      const float bv = bias[gn0 + nc * 16 + lo];
#pragma unroll
      for (int rb = 0; rb < 2; ++rb) {
        v8bf pk;
#pragma unroll
        for (int j = 0; j < 8; ++j) pk[j] = to_bf16(acc[rb][nc][j] + bv);
        *(v8bf*)&Vt[(bh * DD + d) * TT + t0 + rb * 16 + 8 * h] = pk;
      }
    }
  } else {
    bf16* __restrict__ dst = (sel == 0) ? Qb : Kb;
    const float scale = (sel == 0) ? 0.125f : 1.0f;  // fold 1/sqrt(D) into Q
#pragma unroll
    for (int nc = 0; nc < 4; ++nc) {
      const int d = nc * 16 + lo;
      const float bv = bias[gn0 + nc * 16 + lo];
#pragma unroll
      for (int rb = 0; rb < 2; ++rb) {
#pragma unroll
        for (int j = 0; j < 8; ++j) {
          const int t = t0 + rb * 16 + j + 8 * h;
          dst[(bh * TT + t) * DD + d] = to_bf16((acc[rb][nc][j] + bv) * scale);
        }
      }
    }
  }
}

// ---------------------------------------------------------------------------
// Kernel 2: causal flash attention. 4 waves/block, 16 q rows per wave.
// ---------------------------------------------------------------------------
__global__ __launch_bounds__(128) void attn_kernel(
    const bf16* __restrict__ Qb, const bf16* __restrict__ Kb,
    const bf16* __restrict__ Vt, bf16* __restrict__ Ab) {
  __shared__ alignas(16) bf16 Pt[4][16 * 40];  // per-wave P tile
  const int tid = threadIdx.x, wave = tid >> 5, lane = tid & 31;
  const int h = lane >> 4, lo = lane & 15;
  const int bh = blockIdx.y;
  const int b_ = bh >> 4, head = bh & 15;
  const int qbase = blockIdx.x * 64 + wave * 16;

  const bf16* Qh = Qb + (size_t)bh * TT * DD;
  const bf16* Kh = Kb + (size_t)bh * TT * DD;
  const bf16* Vh = Vt + (size_t)bh * DD * TT;  // transposed [D][T]

  const v16bf qf0 = load_frag_a(Qh + (size_t)qbase * DD + 0, DD);
  const v16bf qf1 = load_frag_a(Qh + (size_t)qbase * DD + 32, DD);

  float mst[8], lst[8];
#pragma unroll
  for (int j = 0; j < 8; ++j) { mst[j] = -1e30f; lst[j] = 0.f; }
  v8f o[4] = {};

  bf16* P = &Pt[wave][0];
  const int kend = qbase + 16;
  for (int kb = 0; kb < kend; kb += 32) {
    // --- S = Q K^T ---
    v8f s0 = {}, s1 = {};
    {
      v16bf kf;
      kf = load_frag_bT(Kh + (size_t)(kb + 0) * DD + 0, DD);
      s0 = wmma_bf16(qf0, kf, s0);
      kf = load_frag_bT(Kh + (size_t)(kb + 0) * DD + 32, DD);
      s0 = wmma_bf16(qf1, kf, s0);
      kf = load_frag_bT(Kh + (size_t)(kb + 16) * DD + 0, DD);
      s1 = wmma_bf16(qf0, kf, s1);
      kf = load_frag_bT(Kh + (size_t)(kb + 16) * DD + 32, DD);
      s1 = wmma_bf16(qf1, kf, s1);
    }
    // --- causal mask + online softmax ---
#pragma unroll
    for (int j = 0; j < 8; ++j) {
      int q = qbase + j + 8 * h;
      float v0 = ((kb + lo) <= q) ? s0[j] : -1e30f;
      float v1 = ((kb + 16 + lo) <= q) ? s1[j] : -1e30f;
      float cm = fmaxf(v0, v1);
      cm = fmaxf(cm, __shfl_xor(cm, 1));
      cm = fmaxf(cm, __shfl_xor(cm, 2));
      cm = fmaxf(cm, __shfl_xor(cm, 4));
      cm = fmaxf(cm, __shfl_xor(cm, 8));
      float mnew = fmaxf(mst[j], cm);
      float alpha = __expf(mst[j] - mnew);
      float p0 = __expf(v0 - mnew);
      float p1 = __expf(v1 - mnew);
      float rs = p0 + p1;
      rs += __shfl_xor(rs, 1);
      rs += __shfl_xor(rs, 2);
      rs += __shfl_xor(rs, 4);
      rs += __shfl_xor(rs, 8);
      lst[j] = lst[j] * alpha + rs;
      mst[j] = mnew;
#pragma unroll
      for (int c = 0; c < 4; ++c) o[c][j] *= alpha;
      int row = j + 8 * h;
      P[row * 40 + lo] = to_bf16(p0);
      P[row * 40 + 16 + lo] = to_bf16(p1);
    }
    asm volatile("s_wait_dscnt 0" ::: "memory");  // per-wave LDS RAW fence
    v16bf af = load_frag_a(P, 40);
#pragma unroll
    for (int c = 0; c < 4; ++c) {
      const bf16* row = Vh + (size_t)(c * 16 + lo) * TT + kb + h * 16;
      v16bf vf = join16(ld8(row), ld8(row + 8));
      o[c] = wmma_bf16(af, vf, o[c]);
    }
  }
  // --- normalize + write bf16 attn output [B][T][H*D] ---
#pragma unroll
  for (int j = 0; j < 8; ++j) {
    int q = qbase + j + 8 * h;
    float inv = 1.0f / lst[j];
    size_t base = ((size_t)b_ * TT + q) * CC + head * DD;
#pragma unroll
    for (int c = 0; c < 4; ++c)
      Ab[base + c * 16 + lo] = to_bf16(o[c][j] * inv);
  }
}

// ---------------------------------------------------------------------------
// Kernel 3: out = attn(bf16) @ w_proj + b_proj (f32). A tile staged by TDM.
// ---------------------------------------------------------------------------
__global__ __launch_bounds__(128) void proj_kernel(
    const bf16* __restrict__ A, const float* __restrict__ w,
    const float* __restrict__ bias, float* __restrict__ out) {
  __shared__ alignas(16) bf16 As[128 * 40];
  __shared__ alignas(16) bf16 Bsw[4 * 32 * 24];
  const int tid = threadIdx.x, wave = tid >> 5, lane = tid & 31;
  const int h = lane >> 4, lo = lane & 15;
  const int gn0 = blockIdx.x * 64;
  const int gm0 = blockIdx.y * 128;
  const unsigned lds_As = (unsigned)(uintptr_t)&As[0];  // LDS byte offset

  v8f acc[2][4] = {};
  for (int k0 = 0; k0 < CC; k0 += 32) {
    // A tile 128x32 bf16: one async TDM DMA issued by wave 0 (EXEC ignored),
    // LDS padded to the 40-element row stride via pad_interval/pad_amount.
    if (wave == 0)
      tdm_load_tile_bf16(lds_As, &A[(size_t)gm0 * CC + k0], 128, CC);
    // stage B 32x64 swizzled (f32 -> bf16), overlaps with the TDM transfer
#pragma unroll
    for (int i = 0; i < 4; ++i) {
      int idx = tid + i * 128;
      int r = idx >> 4, c0 = (idx & 15) * 4;
      const float4 f = *(const float4*)&w[(size_t)(k0 + r) * CC + gn0 + c0];
      int base = (((c0 >> 4) * 32) + ((r >> 4) * 16) + (c0 & 15)) * 24 + (r & 15);
      Bsw[base +  0] = to_bf16(f.x);
      Bsw[base + 24] = to_bf16(f.y);
      Bsw[base + 48] = to_bf16(f.z);
      Bsw[base + 72] = to_bf16(f.w);
    }
    if (k0 + 32 < CC)
      __builtin_prefetch(&w[(size_t)(k0 + 32 + (tid >> 2)) * CC + gn0], 0, 3);
    if (wave == 0) __builtin_amdgcn_s_wait_tensorcnt(0);
    __syncthreads();
    v16bf a0 = load_frag_a(As + (wave * 32 + 0) * 40, 40);
    v16bf a1 = load_frag_a(As + (wave * 32 + 16) * 40, 40);
#pragma unroll
    for (int nc = 0; nc < 4; ++nc) {
      v16bf b = load_frag_b_sw(Bsw + nc * 32 * 24);
      acc[0][nc] = wmma_bf16(a0, b, acc[0][nc]);
      acc[1][nc] = wmma_bf16(a1, b, acc[1][nc]);
    }
    __syncthreads();
  }
#pragma unroll
  for (int nc = 0; nc < 4; ++nc) {
    int gn = gn0 + nc * 16 + lo;
    float bv = bias[gn];
#pragma unroll
    for (int rb = 0; rb < 2; ++rb) {
#pragma unroll
      for (int j = 0; j < 8; ++j) {
        int gm = gm0 + wave * 32 + rb * 16 + j + 8 * h;
        out[(size_t)gm * CC + gn] = acc[rb][nc][j] + bv;
      }
    }
  }
}

extern "C" void kernel_launch(void* const* d_in, const int* in_sizes, int n_in,
                              void* d_out, int out_size, void* d_ws,
                              size_t ws_size, hipStream_t stream) {
  const float* x = (const float*)d_in[0];
  const float* w_attn = (const float*)d_in[1];
  const float* b_attn = (const float*)d_in[2];
  const float* w_proj = (const float*)d_in[3];
  const float* b_proj = (const float*)d_in[4];
  float* out = (float*)d_out;

  const size_t per = (size_t)NB * NH * TT * DD;  // 4M bf16 elems each
  bf16* Qb = (bf16*)d_ws;
  bf16* Kb = Qb + per;
  bf16* Vt = Kb + per;   // transposed [B][H][D][T]
  bf16* Ab = Vt + per;   // [B*T][C]

  qkv_kernel<<<dim3(N3 / 64, MM / 128), 128, 0, stream>>>(x, w_attn, b_attn,
                                                          Qb, Kb, Vt);
  attn_kernel<<<dim3(TT / 64, NB * NH), 128, 0, stream>>>(Qb, Kb, Vt, Ab);
  proj_kernel<<<dim3(CC / 64, MM / 128), 128, 0, stream>>>(Ab, w_proj, b_proj,
                                                           out);
}
